// RNN_3006477107833
// MI455X (gfx1250) — compile-verified, matrix operand order checked by reading, not run on hardware
//
#include <hip/hip_runtime.h>
#include <math.h>

typedef __attribute__((ext_vector_type(16))) _Float16 v16h;
typedef __attribute__((ext_vector_type(8)))  float    v8f;

#if __has_builtin(__builtin_amdgcn_tanhf)
#define TANHF(v) __builtin_amdgcn_tanhf(v)
#else
#define TANHF(v) tanhf(v)
#endif

// Problem constants (fixed by the reference).
constexpr int BATCH = 8192;
constexpr int TSTEP = 784;
constexpr int H     = 30;   // hidden, padded to K=32 inside WMMA
constexpr int NC    = 10;   // output classes
constexpr int S     = 34;   // LDS row stride in dwords (padded, even -> 8B pairs)

// Compiler-only reordering fence: same-wave DS ops are processed in order by
// hardware, so no hardware DScnt drain is needed for intra-wave store->load.
#define SW_BARRIER() asm volatile("" ::: "memory")

// One wave owns TWO independent 16-row batch chains (ILP=2 to cover the
// serial WMMA->tanh->LDS->cvt dependency when a wave is alone on a SIMD).
//
// Per chain and per step, one K=32 WMMA pair computes the whole
// pre-activation:
//   A (16x32 f16): A[b][k]=h[k] (k<30), A[b][30]=x_t[b], A[b][31]=1
//   B (32x16 f16): B[k][j]=W_hh[j][k] (k<30), B[30][j]=W_ih[j],
//                  B[31][j]=b_ih[j]+b_hh[j]; columns j>=30 are zero.
__global__ __launch_bounds__(32)
void rnn_scan_kernel(const float* __restrict__ x,     // [B, T]  (I==1)
                     const float* __restrict__ W_ih,  // [H, 1]
                     const float* __restrict__ W_hh,  // [H, H]
                     const float* __restrict__ b_ih,  // [H]
                     const float* __restrict__ b_hh,  // [H]
                     const float* __restrict__ W_fc,  // [C, H]
                     const float* __restrict__ b_fc,  // [C]
                     float* __restrict__ out)         // [B, C]
{
    __shared__ float lds[2][16 * S];  // one h tile per chain

    const int lane = threadIdx.x & 31;
    const int hi   = lane >> 4;        // 0: lanes 0-15, 1: lanes 16-31
    const int nloc = lane & 15;
    const int base = blockIdx.x * 32;  // 32 batch rows per wave (2 chains)
    const int mb   = hi * 8;           // C/D row base for this half-wave

    // Per-lane x row index for each chain (value only consumed by half1).
    const long ix0 = (long)(base + nloc) * TSTEP;
    const long ix1 = ix0 + 16L * TSTEP;

    // K pair base per VGPR slot r (16-bit A/B layout, wave32):
    //  half0: K = {0..7} in v0..v3, {16..23} in v4..v7
    //  half1: K = {8..15} in v0..v3, {24..31} in v4..v7
    int kp[8];
#pragma unroll
    for (int r = 0; r < 8; ++r)
        kp[r] = (r < 4 ? 2 * r : 16 + 2 * (r - 4)) + hi * 8;

    // ---- Build B fragments once (shared by both chains, live in VGPRs) ----
    v16h b0, b1;
#pragma unroll
    for (int r = 0; r < 8; ++r) {
#pragma unroll
        for (int s = 0; s < 2; ++s) {
            const int k  = kp[r] + s;
            const int j0 = nloc;        // columns 0..15  -> tile c0
            const int j1 = nloc + 16;   // columns 16..31 -> tile c1
            float v0 = (k < H) ? W_hh[j0 * H + k]
                               : (k == H ? W_ih[j0] : (b_ih[j0] + b_hh[j0]));
            float v1 = (j1 < H)
                           ? ((k < H) ? W_hh[j1 * H + k]
                                      : (k == H ? W_ih[j1] : (b_ih[j1] + b_hh[j1])))
                           : 0.0f;
            b0[2 * r + s] = (_Float16)v0;
            b1[2 * r + s] = (_Float16)v1;
        }
    }

    // ---- Initial A for both chains: h0 = 0, K slots {30,31} = (x_0, 1) ----
    v16h A[2];
    {
        const float x00 = x[ix0];
        const float x01 = x[ix1];
#pragma unroll
        for (int i = 0; i < 16; ++i) { A[0][i] = (_Float16)0.0f; A[1][i] = (_Float16)0.0f; }
        A[0][14] = (_Float16)(hi ? x00  : 0.0f);
        A[0][15] = (_Float16)(hi ? 1.0f : 0.0f);
        A[1][14] = (_Float16)(hi ? x01  : 0.0f);
        A[1][15] = (_Float16)(hi ? 1.0f : 0.0f);
    }

    // ---- Sequential scan: T-1 full steps + peeled final step ----
    for (int t = 0; t < TSTEP - 1; ++t) {
        // Next-step x loads issue first so they overlap the whole body.
        const float xn0 = x[ix0 + t + 1];
        const float xn1 = x[ix1 + t + 1];

        v8f c0[2], c1[2];
#pragma unroll
        for (int u = 0; u < 2; ++u) {
            v8f z = {};
            c0[u] = __builtin_amdgcn_wmma_f32_16x16x32_f16(false, A[u], false, b0,
                                                           (short)0, z, false, false);
            c1[u] = __builtin_amdgcn_wmma_f32_16x16x32_f16(false, A[u], false, b1,
                                                           (short)0, z, false, false);
        }

        // h = tanh(D) -> LDS in [row m][col j] layout, per chain.
#pragma unroll
        for (int u = 0; u < 2; ++u)
#pragma unroll
            for (int r = 0; r < 8; ++r) {
                lds[u][(mb + r) * S + nloc]      = TANHF(c0[u][r]);
                lds[u][(mb + r) * S + nloc + 16] = TANHF(c1[u][r]);
            }
        SW_BARRIER();   // keep loads after stores; in-order LDS does the rest

        // Rebuild A fragments for the next step (transpose via LDS).
#pragma unroll
        for (int u = 0; u < 2; ++u) {
            const float xn = u ? xn1 : xn0;
#pragma unroll
            for (int r = 0; r < 8; ++r) {
                const int  k   = kp[r];
                const bool isx = (hi == 1) && (r == 7);  // K pair {30,31}
                const float h0 = lds[u][nloc * S + k];
                const float h1 = lds[u][nloc * S + k + 1];
                A[u][2 * r]     = (_Float16)(isx ? xn   : h0);
                A[u][2 * r + 1] = (_Float16)(isx ? 1.0f : h1);
            }
        }
        SW_BARRIER();   // next iter's stores stay after this iter's loads
    }

    // ---- Final step (no A rebuild needed) ----
    {
        v8f c0[2], c1[2];
#pragma unroll
        for (int u = 0; u < 2; ++u) {
            v8f z = {};
            c0[u] = __builtin_amdgcn_wmma_f32_16x16x32_f16(false, A[u], false, b0,
                                                           (short)0, z, false, false);
            c1[u] = __builtin_amdgcn_wmma_f32_16x16x32_f16(false, A[u], false, b1,
                                                           (short)0, z, false, false);
        }
#pragma unroll
        for (int u = 0; u < 2; ++u)
#pragma unroll
            for (int r = 0; r < 8; ++r) {
                lds[u][(mb + r) * S + nloc]      = TANHF(c0[u][r]);
                lds[u][(mb + r) * S + nloc + 16] = TANHF(c1[u][r]);
            }
        SW_BARRIER();
    }

    // ---- Epilogue: out = h_T @ W_fc^T + b_fc.  Half-wave u=hi handles
    //      chain hi, row m=nloc; both halves fully busy. ----
    {
        const int u = hi;
        const int m = nloc;
#pragma unroll
        for (int c = 0; c < NC; ++c) {
            float acc = b_fc[c];
#pragma unroll
            for (int k = 0; k < H; ++k)
                acc += lds[u][m * S + k] * W_fc[c * H + k];
            out[(base + u * 16 + m) * NC + c] = acc;
        }
    }
}

extern "C" void kernel_launch(void* const* d_in, const int* in_sizes, int n_in,
                              void* d_out, int out_size, void* d_ws, size_t ws_size,
                              hipStream_t stream)
{
    (void)in_sizes; (void)n_in; (void)out_size; (void)d_ws; (void)ws_size;
    const float* x    = (const float*)d_in[0];
    const float* W_ih = (const float*)d_in[1];
    const float* W_hh = (const float*)d_in[2];
    const float* b_ih = (const float*)d_in[3];
    const float* b_hh = (const float*)d_in[4];
    const float* W_fc = (const float*)d_in[5];
    const float* b_fc = (const float*)d_in[6];
    float* out = (float*)d_out;

    dim3 grid(BATCH / 32);  // 256 waves, two 16-row chains each
    dim3 block(32);
    rnn_scan_kernel<<<grid, block, 0, stream>>>(x, W_ih, W_hh, b_ih, b_hh,
                                                W_fc, b_fc, out);
}
